// PointNet_Plus_1760936592067
// MI455X (gfx1250) — matrix operand order repaired and checked by reading, not compiled, not run on hardware
//
#include <hip/hip_runtime.h>
#include <hip/hip_bf16.h>
#include <cstddef>
#include <cstdint>

// ---------------------------------------------------------------------------
// PointNet++ head for MI455X (gfx1250, wave32, WMMA + async-to-LDS DMA).
// Every MLP layer is a WMMA GEMM (f16 A/B, f32 accumulate) with fused
// bias + BN-scale + ReLU epilogue. Regular layers use a double-buffered
// global_load_async_to_lds pipeline (ASYNCcnt) so DMA overlaps WMMA.
// kNN gather is fused into gather-mode B staging (no 225 MB grouped tensor).
// ---------------------------------------------------------------------------

typedef __attribute__((ext_vector_type(16))) _Float16 v16h;
typedef __attribute__((ext_vector_type(8)))  float    v8f;

#define BN_INVF 0.99999500003749978f   // 1/sqrt(1 + 1e-5)
#define FLT_BIG 3.402823466e+38f

enum { BB = 8, S1 = 512, S2 = 128, KK = 64, N0 = 1024, FJ = 84 };

// ---------------------------------------------------------------------------
// WMMA GEMM. Out[b,o,n] = relu(g[o]*((sum_c W[o,c]*val(b,c,n) + bias[o])*BN_INV) + beta[o])
//   GATHER=false: val = In[(b*Cin+c)*inStride + n]
//   GATHER=true : n -> center j=n>>6, slot kk=n&63, src=Idx[(b*(N/64)+j)*64+kk]
//                 val = Pts[..src] - (c<3 ? Pts[..j] : 0)
// 128 threads = 4 wave32 waves. Block tile 16(Cout) x 128(N); each wave owns a
// 16x32 slab (2 accumulators sharing one A fragment). K chunks of 32,
// double-buffered LDS (f32 tiles; f32->f16 at fragment build).
// ---------------------------------------------------------------------------
template <bool GATHER>
__global__ __launch_bounds__(128) void gemm_bn_relu(
    const float* __restrict__ W, const float* __restrict__ bias,
    const float* __restrict__ gamma, const float* __restrict__ beta,
    const float* __restrict__ In, const int* __restrict__ Idx,
    int inStride, int Cin, int Cout, int N, float* __restrict__ Out)
{
    constexpr int APF = 36;    // A row stride (f32)
    constexpr int BPF = 132;   // B row stride (f32), 528 B (16B-aligned)
    __shared__ __align__(16) float lAf[2][16 * APF];
    __shared__ __align__(16) float lBf[2][32 * BPF];

    const int tid  = threadIdx.x;
    const int lane = tid & 31;
    const int wid  = tid >> 5;
    const int b     = blockIdx.z;
    const int oBase = blockIdx.y * 16;
    const int nBase = blockIdx.x * 128;

    v8f acc0 = {}, acc1 = {};

    // ---- async DMA staging of one full K-chunk (B: 8x b128, A: 4x b32) ----
    auto issueAsync = [&](int k0, int sel) {
        #pragma unroll
        for (int r = 0; r < 8; ++r) {
            int q   = tid + r * 128;     // 0..1023 b128 chunks of B
            int row = q >> 5;            // K row 0..31
            int c4  = (q & 31) << 2;     // dword col 0,4,..,124
            const float* gp =
                In + ((size_t)b * Cin + (k0 + row)) * inStride + nBase + c4;
            unsigned lp = (unsigned)(uintptr_t)(&lBf[sel][row * BPF + c4]);
            asm volatile("global_load_async_to_lds_b128 %0, %1, off"
                         :: "v"(lp), "v"((uint64_t)(uintptr_t)gp) : "memory");
        }
        #pragma unroll
        for (int r = 0; r < 4; ++r) {
            int e = tid + r * 128;       // 0..511 dwords of A
            int o = e >> 5, k = e & 31;
            const float* gp = W + (size_t)(oBase + o) * Cin + k0 + k;
            unsigned lp = (unsigned)(uintptr_t)(&lAf[sel][o * APF + k]);
            asm volatile("global_load_async_to_lds_b32 %0, %1, off"
                         :: "v"(lp), "v"((uint64_t)(uintptr_t)gp) : "memory");
        }
    };

    // ---- compute staging for partial chunks / gather. multiply-by-flag keeps
    //      loads unconditional (x*0 not foldable => no exec predication). ----
    auto stageManual = [&](int k0, int sel) {
        #pragma unroll
        for (int r = 0; r < 4; ++r) {
            int e = tid + r * 128;
            int o = e >> 5, k = e & 31;
            int oo = oBase + o, c = k0 + k;
            int ooC = oo < Cout ? oo : (Cout - 1);
            int cC  = c  < Cin  ? c  : (Cin - 1);
            float fm = (oo < Cout && c < Cin) ? 1.0f : 0.0f;
            lAf[sel][o * APF + k] = W[(size_t)ooC * Cin + cC] * fm;
        }
        #pragma unroll 4
        for (int r = 0; r < 32; ++r) {
            int e = tid + r * 128;       // 0..4095
            int k = e >> 7, n = e & 127;
            int c = k0 + k, nn = nBase + n;
            int cC = c < Cin ? c : (Cin - 1);
            float fc = (c < Cin) ? 1.0f : 0.0f;
            const float* row = In + ((size_t)b * Cin + cC) * inStride;
            float v;
            if (GATHER) {
                int j  = nn >> 6;
                int kk = nn & 63;
                int src = Idx[((size_t)b * (N >> 6) + j) * 64 + kk];
                float f3 = (c < 3) ? 1.0f : 0.0f;
                v = row[src] - row[j] * f3;          // xyz rel. center
            } else {
                v = row[nn];
            }
            lBf[sel][k * BPF + n] = v * fc;
        }
    };

    // ---- fragments per ISA 7.12.2 layouts; one A frag feeds two WMMAs ----
    auto compute = [&](int sel) {
        v16h af, bf0, bf1;
        const int mA    = lane & 15;
        const int halfA = lane >> 4;
        const int nB    = lane & 15;
        const int nSub  = wid * 32;
        #pragma unroll
        for (int i = 0; i < 16; ++i) {
            int vg = i >> 1, sl = i & 1;
            int kA = ((vg < 4) ? 0 : 16) + ((vg & 3) * 2 + sl) + (halfA << 3);
            af[i]  = (_Float16)lAf[sel][mA * APF + kA];
            bf0[i] = (_Float16)lBf[sel][kA * BPF + nSub + nB];
            bf1[i] = (_Float16)lBf[sel][kA * BPF + nSub + 16 + nB];
        }
        acc0 = __builtin_amdgcn_wmma_f32_16x16x32_f16(
            false, af, false, bf0, (short)0, acc0, false, false);
        acc1 = __builtin_amdgcn_wmma_f32_16x16x32_f16(
            false, af, false, bf1, (short)0, acc1, false, false);
    };

    const bool regular = !GATHER && ((Cin & 31) == 0) && (oBase + 16 <= Cout);
    if (regular) {
        const int nc = Cin >> 5;
        issueAsync(0, 0);
        for (int ic = 0; ic < nc; ++ic) {
            if (ic + 1 < nc) {
                issueAsync((ic + 1) << 5, (ic + 1) & 1);
                asm volatile("s_wait_asynccnt 0xc" ::: "memory"); // older chunk done
            } else {
                asm volatile("s_wait_asynccnt 0x0" ::: "memory");
            }
            __syncthreads();
            compute(ic & 1);
            __syncthreads();   // all waves done reading before buffer reuse
        }
    } else {
        int sel = 0;
        for (int k0 = 0; k0 < Cin; k0 += 32, sel ^= 1) {
            stageManual(k0, sel);
            __syncthreads();
            compute(sel);
            __syncthreads();
        }
    }

    // ---- epilogue: C/D layout n = lane&15, m = r + 8*(lane>=16) ----
    const int half = lane >> 4, nl = lane & 15;
    const int nn0 = nBase + wid * 32 + nl;
    #pragma unroll
    for (int r = 0; r < 8; ++r) {
        int m = r + (half << 3);
        int o = oBase + m;
        if (o < Cout) {
            float* orow = Out + ((size_t)b * Cout + o) * N;
            float v0 = gamma[o] * ((acc0[r] + bias[o]) * BN_INVF) + beta[o];
            float v1 = gamma[o] * ((acc1[r] + bias[o]) * BN_INVF) + beta[o];
            orow[nn0]      = fmaxf(v0, 0.0f);
            orow[nn0 + 16] = fmaxf(v1, 0.0f);
        }
    }
}

// ---------------------------------------------------------------------------
// Row-wise max over last dim L (contiguous). rows = B*C*S.
// ---------------------------------------------------------------------------
__global__ void rowmax(const float* __restrict__ in, float* __restrict__ out,
                       int rows, int L)
{
    int r = blockIdx.x * blockDim.x + threadIdx.x;
    if (r >= rows) return;
    const float* p = in + (size_t)r * L;
    float m = p[0];
    for (int i = 1; i < L; ++i) m = fmaxf(m, p[i]);
    out[r] = m;
}

// ---------------------------------------------------------------------------
// kNN top-64 selection per (b, center). pts (B, Ctot, stride), xyz ch 0..2;
// centers = first ncent points. Ties -> lowest index. d > r2 -> idx = center.
// ---------------------------------------------------------------------------
__global__ void knn_select(const float* __restrict__ pts, int Ctot, int stride,
                           int npts, int ncent, float r2,
                           int* __restrict__ idx, int Bn)
{
    int t = blockIdx.x * blockDim.x + threadIdx.x;
    if (t >= Bn * ncent) return;
    int b = t / ncent, j = t - b * ncent;
    const float* px = pts + (size_t)b * Ctot * stride;
    float cx = px[j], cy = px[stride + j], cz = px[2 * stride + j];
    unsigned used[32];
    for (int i = 0; i < 32; ++i) used[i] = 0u;
    int* od = idx + (size_t)t * KK;
    for (int k = 0; k < KK; ++k) {
        float bd = FLT_BIG; int bi = 0;
        for (int i = 0; i < npts; ++i) {
            if ((used[i >> 5] >> (i & 31)) & 1u) continue;
            float dx = px[i] - cx, dy = px[stride + i] - cy, dz = px[2 * stride + i] - cz;
            float d = dx * dx + dy * dy + dz * dz;
            if (d < bd) { bd = d; bi = i; }
        }
        used[bi >> 5] |= 1u << (bi & 31);
        od[k] = (bd > r2) ? j : bi;
    }
}

// ---------------------------------------------------------------------------
// 3-NN with inverse-distance weights (pointnetfc interpolation branch).
// ---------------------------------------------------------------------------
__global__ void knn3_idx(const float* __restrict__ tgt, int tCtot, int tStride,
                         const float* __restrict__ src, int sCtot, int sStride, int Ns,
                         int Nt, int Bn, int* __restrict__ idx3, float* __restrict__ w3)
{
    int t = blockIdx.x * blockDim.x + threadIdx.x;
    if (t >= Bn * Nt) return;
    int b = t / Nt, n = t - b * Nt;
    const float* tb = tgt + (size_t)b * tCtot * tStride;
    float tx = tb[n], ty = tb[tStride + n], tz = tb[2 * tStride + n];
    const float* sb = src + (size_t)b * sCtot * sStride;
    float d0 = FLT_BIG, d1 = FLT_BIG, d2 = FLT_BIG;
    int i0 = 0, i1 = 0, i2 = 0;
    for (int i = 0; i < Ns; ++i) {
        float dx = tx - sb[i], dy = ty - sb[sStride + i], dz = tz - sb[2 * sStride + i];
        float d = dx * dx + dy * dy + dz * dz;
        if (d < d0)      { d2 = d1; i2 = i1; d1 = d0; i1 = i0; d0 = d; i0 = i; }
        else if (d < d1) { d2 = d1; i2 = i1; d1 = d;  i1 = i; }
        else if (d < d2) { d2 = d;  i2 = i; }
    }
    d0 = fmaxf(d0, 1e-10f); d1 = fmaxf(d1, 1e-10f); d2 = fmaxf(d2, 1e-10f);
    float w0 = 1.f / d0, w1 = 1.f / d1, w2 = 1.f / d2;
    float s = w0 + w1 + w2;
    idx3[t * 3 + 0] = i0; idx3[t * 3 + 1] = i1; idx3[t * 3 + 2] = i2;
    w3[t * 3 + 0] = w0 / s; w3[t * 3 + 1] = w1 / s; w3[t * 3 + 2] = w2 / s;
}

__global__ void interp_apply(float* __restrict__ dst, int dstCtot, int c0, int dN,
                             const float* __restrict__ p2, int pCtot, int pStride,
                             const int* __restrict__ idx3, const float* __restrict__ w3,
                             int C, int Nt, int Bn)
{
    int t = blockIdx.x * blockDim.x + threadIdx.x;
    int tot = Bn * C * Nt;
    if (t >= tot) return;
    int n = t % Nt; int rc = t / Nt; int c = rc % C; int b = rc / C;
    const int*   id = idx3 + ((size_t)b * Nt + n) * 3;
    const float* w  = w3   + ((size_t)b * Nt + n) * 3;
    const float* row = p2 + ((size_t)b * pCtot + c) * pStride;
    float v = w[0] * row[id[0]] + w[1] * row[id[1]] + w[2] * row[id[2]];
    dst[((size_t)b * dstCtot + c0 + c) * dN + n] = v;
}

// ---------------------------------------------------------------------------
// Channel copy / broadcast helpers (concat building blocks).
// ---------------------------------------------------------------------------
__global__ void copy_ch(float* __restrict__ dst, int dstCtot, int dstN, int c0,
                        const float* __restrict__ src, int srcCtot, int srcStride, int csrc0,
                        int C, int N, int Bn)
{
    int t = blockIdx.x * blockDim.x + threadIdx.x;
    int tot = Bn * C * N;
    if (t >= tot) return;
    int n = t % N; int rc = t / N; int c = rc % C; int b = rc / C;
    dst[((size_t)b * dstCtot + c0 + c) * dstN + n] =
        src[((size_t)b * srcCtot + csrc0 + c) * srcStride + n];
}

__global__ void bcast_ch(float* __restrict__ dst, int dstCtot, int c0, int dN,
                         const float* __restrict__ src, int C, int Bn)
{
    int t = blockIdx.x * blockDim.x + threadIdx.x;
    int tot = Bn * C * dN;
    if (t >= tot) return;
    int n = t % dN; int rc = t / dN; int c = rc % C; int b = rc / C;
    dst[((size_t)b * dstCtot + c0 + c) * dN + n] = src[(size_t)b * C + c];
}

// ---------------------------------------------------------------------------
// Host orchestration
// ---------------------------------------------------------------------------
struct Layer { const float *W, *b, *g, *be; int cin, cout; };

extern "C" void kernel_launch(void* const* d_in, const int* in_sizes, int n_in,
                              void* d_out, int out_size, void* d_ws, size_t ws_size,
                              hipStream_t stream) {
    (void)in_sizes; (void)n_in; (void)out_size; (void)ws_size;

    const float* xin = (const float*)d_in[0];   // (8,3,512,64)
    const float* y   = (const float*)d_in[1];   // (8,3,512)
    const float* x00 = (const float*)d_in[2];   // (8,3,1024)

    auto L = [&](int i, int cin, int cout) {
        return Layer{ (const float*)d_in[i], (const float*)d_in[i+1],
                      (const float*)d_in[i+2], (const float*)d_in[i+3], cin, cout };
    };
    Layer net1[3]  = { L(3,3,64),    L(7,64,64),   L(11,64,128) };
    Layer net11[3] = { L(15,215,64), L(19,64,64),  L(23,64,128) };
    Layer net2[3]  = { L(27,131,128),L(31,128,128),L(35,128,256) };
    Layer net3[3]  = { L(39,259,256),L(43,256,512),L(47,512,1024) };
    Layer net4[2]  = { L(51,1280,256), L(55,256,256) };
    Layer net5[2]  = { L(59,384,256),  L(63,256,128) };
    Layer net6[3]  = { L(67,128,128), L(71,128,128), L(75,128,128) };
    Layer fc1      = L(79,128,84);

    // ---- workspace layout (f32 units) ----
    float* wsf = (float*)d_ws;
    size_t off = 0;
    auto alloc = [&](size_t n) { float* p = wsf + off; off += n; return p; };
    float* bufA   = alloc(16777216);             // 8*64*32768
    float* bufB   = alloc(16777216);
    float* bufC   = alloc(33554432);             // 8*128*32768
    float* x10buf = alloc((size_t)BB*128*S1);
    float* x11    = alloc((size_t)BB*131*S1);
    int*   idx2   = (int*)alloc((size_t)BB*S2*KK);
    float* x20    = alloc((size_t)BB*256*S2);
    float* x21    = alloc((size_t)BB*259*S2);
    float* t256   = alloc((size_t)BB*256*S2);
    float* t512   = alloc((size_t)BB*512*S2);
    float* t1024  = alloc((size_t)BB*1024*S2);
    float* x30    = alloc((size_t)BB*1024);
    float* x20fin = alloc((size_t)BB*1280*S2);
    float* t256b  = alloc((size_t)BB*256*S2);
    int*   idx3a  = (int*)alloc((size_t)BB*S1*3);
    float* w3a    = alloc((size_t)BB*S1*3);
    float* x10f   = alloc((size_t)BB*384*S1);
    float* t256c  = alloc((size_t)BB*256*S1);
    float* x10s1  = alloc((size_t)BB*128*S1);
    float* x10s2  = alloc((size_t)BB*128*S1);
    int*   idx3b  = (int*)alloc((size_t)BB*N0*3);
    float* w3b    = alloc((size_t)BB*N0*3);
    float* x01in  = alloc((size_t)BB*128*N0);
    float* t128a  = alloc((size_t)BB*128*N0);
    float* t128b  = alloc((size_t)BB*128*N0);
    float* x01    = alloc((size_t)BB*128*N0);
    float* xg     = alloc((size_t)BB*215*N0);
    int*   idxg   = (int*)alloc((size_t)BB*S1*KK);

    float* hm1 = (float*)d_out;
    float* hm2 = (float*)d_out + (size_t)BB * FJ * N0;

    // ---- launch helpers ----
    auto gemm = [&](const Layer& ly, const float* In, const int* Idx,
                    int inStride, int N, float* Out) {
        dim3 g(N / 128, (ly.cout + 15) / 16, BB);
        if (Idx)
            gemm_bn_relu<true><<<g, dim3(128), 0, stream>>>(
                ly.W, ly.b, ly.g, ly.be, In, Idx, inStride, ly.cin, ly.cout, N, Out);
        else
            gemm_bn_relu<false><<<g, dim3(128), 0, stream>>>(
                ly.W, ly.b, ly.g, ly.be, In, Idx, inStride, ly.cin, ly.cout, N, Out);
    };
    auto rmax = [&](const float* in, float* out, int rows, int Ld) {
        rowmax<<<dim3((rows + 255) / 256), dim3(256), 0, stream>>>(in, out, rows, Ld);
    };
    auto copy = [&](float* dst, int dstCtot, int dstN, int c0,
                    const float* src, int srcCtot, int srcStride, int csrc0,
                    int C, int N) {
        int tot = BB * C * N;
        copy_ch<<<dim3((tot + 255) / 256), dim3(256), 0, stream>>>(
            dst, dstCtot, dstN, c0, src, srcCtot, srcStride, csrc0, C, N, BB);
    };

    // ---- stage() : shared SA/FP pipeline, used twice ----
    auto run_stage = [&](const float* firstIn, const int* firstIdx,
                         int firstStride, const Layer* f1, float* x10s_out) {
        const int NG = S1 * KK;   // 32768
        gemm(f1[0], firstIn, firstIdx, firstStride, NG, bufA);
        gemm(f1[1], bufA, nullptr, NG, NG, bufB);
        gemm(f1[2], bufB, nullptr, NG, NG, bufC);
        rmax(bufC, x10buf, BB * 128 * S1, KK);                 // x10 (B,128,512)

        copy(x11, 131, S1, 0, y, 3, S1, 0, 3, S1);             // x11 = [y ; x10]
        copy(x11, 131, S1, 3, x10buf, 128, S1, 0, 128, S1);

        knn_select<<<dim3((BB * S2 + 63) / 64), dim3(64), 0, stream>>>(
            x11, 131, S1, S1, S2, 0.08f, idx2, BB);

        const int NG2 = S2 * KK;  // 8192 ; gather fused into GEMM
        gemm(net2[0], x11, idx2, S1, NG2, bufA);
        gemm(net2[1], bufA, nullptr, NG2, NG2, bufB);
        gemm(net2[2], bufB, nullptr, NG2, NG2, bufC);
        rmax(bufC, x20, BB * 256 * S2, KK);                    // x20 (B,256,128)

        copy(x21, 259, S2, 0, x11, 131, S1, 0, 3, S2);         // il2c xyz
        copy(x21, 259, S2, 3, x20, 256, S2, 0, 256, S2);
        gemm(net3[0], x21, nullptr, S2, S2, t256);
        gemm(net3[1], t256, nullptr, S2, S2, t512);
        gemm(net3[2], t512, nullptr, S2, S2, t1024);
        rmax(t1024, x30, BB * 1024, S2);                       // x30 (B,1024)

        copy(x20fin, 1280, S2, 0, x20, 256, S2, 0, 256, S2);   // [x20 ; bcast x30]
        {
            int tot = BB * 1024 * S2;
            bcast_ch<<<dim3((tot + 255) / 256), dim3(256), 0, stream>>>(
                x20fin, 1280, 256, S2, x30, 1024, BB);
        }
        gemm(net4[0], x20fin, nullptr, S2, S2, t256);
        gemm(net4[1], t256, nullptr, S2, S2, t256b);           // x20f (B,256,128)

        knn3_idx<<<dim3((BB * S1 + 255) / 256), dim3(256), 0, stream>>>(
            y, 3, S1, x11, 131, S1, S2, S1, BB, idx3a, w3a);
        copy(x10f, 384, S1, 0, x10buf, 128, S1, 0, 128, S1);   // [x10 ; interp]
        {
            int tot = BB * 256 * S1;
            interp_apply<<<dim3((tot + 255) / 256), dim3(256), 0, stream>>>(
                x10f, 384, 128, S1, t256b, 256, S2, idx3a, w3a, 256, S1, BB);
        }
        gemm(net5[0], x10f, nullptr, S1, S1, t256c);
        gemm(net5[1], t256c, nullptr, S1, S1, x10s_out);       // (B,128,512)
    };

    // ===================== top level =====================
    run_stage(xin, nullptr, S1 * KK, net1, x10s1);

    // x01 = 3-NN interp of x10s1 from y(512) onto x00(1024)
    knn3_idx<<<dim3((BB * N0 + 255) / 256), dim3(256), 0, stream>>>(
        x00, 3, N0, y, 3, S1, S1, N0, BB, idx3b, w3b);
    {
        int tot = BB * 128 * N0;
        interp_apply<<<dim3((tot + 255) / 256), dim3(256), 0, stream>>>(
            x01in, 128, 0, N0, x10s1, 128, S1, idx3b, w3b, 128, N0, BB);
    }
    gemm(net6[0], x01in, nullptr, N0, N0, t128a);
    gemm(net6[1], t128a, nullptr, N0, N0, t128b);
    gemm(net6[2], t128b, nullptr, N0, N0, x01);
    gemm(fc1, x01, nullptr, N0, N0, hm1);                      // heatmap_1 -> d_out

    // xg = [x00 ; heatmap_1 ; x01]  (B,215,1024)
    copy(xg, 215, N0, 0,  x00, 3,  N0, 0, 3,  N0);
    copy(xg, 215, N0, 3,  hm1, FJ, N0, 0, FJ, N0);
    copy(xg, 215, N0, 87, x01, 128, N0, 0, 128, N0);

    knn_select<<<dim3((BB * S1 + 63) / 64), dim3(64), 0, stream>>>(
        xg, 215, N0, N0, S1, 0.16f, idxg, BB);

    run_stage(xg, idxg, N0, net11, x10s2);                     // gather-mode first MLP

    // x02 path: same 3-NN (x00 vs y) weights as x01 path
    {
        int tot = BB * 128 * N0;
        interp_apply<<<dim3((tot + 255) / 256), dim3(256), 0, stream>>>(
            x01in, 128, 0, N0, x10s2, 128, S1, idx3b, w3b, 128, N0, BB);
    }
    gemm(net6[0], x01in, nullptr, N0, N0, t128a);
    gemm(net6[1], t128a, nullptr, N0, N0, t128b);
    gemm(net6[2], t128b, nullptr, N0, N0, x01);
    gemm(fc1, x01, nullptr, N0, N0, hm2);                      // heatmap_2 -> d_out
}
